// GNNBaseline_34711925686441
// MI455X (gfx1250) — matrix operand order repaired and checked by reading, not compiled, not run on hardware
//
#include <hip/hip_runtime.h>
#include <hip/hip_bf16.h>
#include <math.h>

// ---------------------------------------------------------------------------
// GATv2 GNN forward for MI455X (gfx1250, wave32).
//
// Input flattening assumption (setup_inputs dict order; 'params' flattened as
// a JAX pytree: dicts in sorted-key order, lists in order):
//  0 features[4096*33]          (UNUSED by reference)
//  1 ethnicity[4096] i32
//  2 node_idx[4096] i32
//  3 full_graph_census[20000*32]
//  4 full_graph_population_t[20000*10]
//  5 edge_index[2*320000] i32
//  6 enc_W1[33*128]   7 enc_W2[128*128]  8 enc_b1[128]  9 enc_b2[128]
// 10 enc_ln1_b 11 enc_ln1_g 12 enc_ln2_b 13 enc_ln2_g
// 14 eth_emb[10*32]
// 15+6l: gat[l] = { Wl, Wr, att[4*64], bias, bl, br }   l = 0..2
// 33+2l: norms[l] = { b, g }
// 39 pred_W1[96*128] 40 pred_W2[128*64] 41 pred_W3[64*1]
// 42 pred_b1 43 pred_b2 44 pred_b3
// 45 pred_ln1_b 46 pred_ln1_g 47 pred_ln2_b 48 pred_ln2_g
// ---------------------------------------------------------------------------

#define N_NODES 20000
#define N_EDGES 320000
#define ET_EDGES (N_EDGES + N_NODES)   // + self loops
#define BATCH 4096
#define FE 128
#define HC 256                          // HEADS*HID
#define HID 64
#define PH 128

typedef __attribute__((ext_vector_type(16))) _Float16 v16h;
typedef __attribute__((ext_vector_type(8)))  _Float16 v8h;
typedef __attribute__((ext_vector_type(8)))  float    v8f;

// ---- helpers --------------------------------------------------------------

__device__ __forceinline__ int fkey(float f) {           // order-preserving float->int
  int i = __float_as_int(f);
  return i ^ ((i >> 31) & 0x7fffffff);
}
__device__ __forceinline__ float keyToFloat(int k) {     // involution
  int i = k ^ ((k >> 31) & 0x7fffffff);
  return __int_as_float(i);
}

__device__ __forceinline__ float block_reduce_sum(float v, float* sbuf) {
  int t = threadIdx.x;
  sbuf[t] = v;
  __syncthreads();
  for (int s = blockDim.x >> 1; s > 0; s >>= 1) {
    if (t < s) sbuf[t] += sbuf[t + s];
    __syncthreads();
  }
  float r = sbuf[0];
  __syncthreads();
  return r;
}

// ---- fills ----------------------------------------------------------------

__global__ void fill_f32(float* p, float v, int n) {
  int i = blockIdx.x * blockDim.x + threadIdx.x;
  if (i < n) p[i] = v;
}
__global__ void fill_i32(int* p, int v, int n) {
  int i = blockIdx.x * blockDim.x + threadIdx.x;
  if (i < n) p[i] = v;
}

// ---- encoder: per node, [33]->128->128 with LN+ReLU -----------------------

__global__ __launch_bounds__(128)
void encoder_kernel(const float* __restrict__ census, const float* __restrict__ pop,
                    const float* __restrict__ W1, const float* __restrict__ b1,
                    const float* __restrict__ g1, const float* __restrict__ bb1,
                    const float* __restrict__ W2, const float* __restrict__ b2,
                    const float* __restrict__ g2, const float* __restrict__ bb2,
                    float* __restrict__ H0) {
  __shared__ float sff[33];
  __shared__ float sh[128];
  __shared__ float sbuf[128];
  int n = blockIdx.x, t = threadIdx.x;
  if (t < 32) sff[t] = census[(size_t)n * 32 + t];
  if (t == 32) {
    float s = 0.f;
    for (int i = 0; i < 10; ++i) s += pop[(size_t)n * 10 + i];
    sff[32] = s;
  }
  __syncthreads();

  float acc = b1[t];
  for (int k = 0; k < 33; ++k) acc += sff[k] * W1[k * 128 + t];
  float mean = block_reduce_sum(acc, sbuf) * (1.f / 128);
  float d = acc - mean;
  float var = block_reduce_sum(d * d, sbuf) * (1.f / 128);
  float y = d * rsqrtf(var + 1e-5f) * g1[t] + bb1[t];
  y = y > 0.f ? y : 0.f;
  sh[t] = y;
  __syncthreads();

  float acc2 = b2[t];
  for (int k = 0; k < 128; ++k) acc2 += sh[k] * W2[k * 128 + t];
  float mean2 = block_reduce_sum(acc2, sbuf) * (1.f / 128);
  float d2 = acc2 - mean2;
  float var2 = block_reduce_sum(d2 * d2, sbuf) * (1.f / 128);
  float y2 = d2 * rsqrtf(var2 + 1e-5f) * g2[t] + bb2[t];
  y2 = y2 > 0.f ? y2 : 0.f;
  H0[(size_t)n * 128 + t] = y2;
}

// ---- f32 -> f16 elementwise ----------------------------------------------

__global__ void h_to_half(const float* __restrict__ src, _Float16* __restrict__ dst, int n) {
  int i = blockIdx.x * blockDim.x + threadIdx.x;
  if (i < n) dst[i] = (_Float16)src[i];
}

// ---- weight transpose f32 [Cin x 256] -> f16 [256 x Cin] ------------------

__global__ __launch_bounds__(256)
void transpose_w_f16(const float* __restrict__ W, _Float16* __restrict__ Wt, int Cin) {
  int n = blockIdx.x;   // output column 0..255
  for (int k = threadIdx.x; k < Cin; k += blockDim.x)
    Wt[(size_t)n * Cin + k] = (_Float16)W[(size_t)k * 256 + n];
}

// ---- WMMA GEMM: X = H[N x Cin] @ W[Cin x 256] + bias ----------------------
// grid (1250, 2): x = 16-row tile, y selects (Wl,bl,XL) or (Wr,br,XR).
// block 256 = 8 waves; wave w computes columns [32w,32w+16) and [32w+16,32w+32).

__global__ __launch_bounds__(256)
void gat_gemm_f16(const _Float16* __restrict__ Hh, int Cin,
                  const _Float16* __restrict__ WtL, const _Float16* __restrict__ WtR,
                  const float* __restrict__ bL, const float* __restrict__ bR,
                  float* __restrict__ XL, float* __restrict__ XR) {
  const _Float16* Wt = blockIdx.y ? WtR : WtL;
  const float* bias = blockIdx.y ? bR : bL;
  float* X = blockIdx.y ? XR : XL;

  int tid = threadIdx.x;
  int wave = tid >> 5;
  int lane = tid & 31;
  int half = lane >> 4;
  int l16 = lane & 15;

  int M0 = blockIdx.x * 16;
  int n0 = wave * 32 + l16;
  int n1 = n0 + 16;

  const _Float16* Arow = Hh + (size_t)(M0 + l16) * Cin;
  const _Float16* B0 = Wt + (size_t)n0 * Cin;
  const _Float16* B1 = Wt + (size_t)n1 * Cin;

  float bv0 = bias[n0], bv1 = bias[n1];
  v8f c0, c1;
#pragma unroll
  for (int r = 0; r < 8; ++r) { c0[r] = bv0; c1[r] = bv1; }

  for (int kk = 0; kk < Cin; kk += 32) {
    union { v16h v; v8h h[2]; } a, b0v, b1v;
    // A layout (16-bit, 16x32): lane half 0 -> K kk+[0..8)/kk+16+[0..8);
    //                           lane half 1 -> K kk+8+[0..8)/kk+24+[0..8)
    a.h[0] = *(const v8h*)(Arow + kk + half * 8);
    a.h[1] = *(const v8h*)(Arow + kk + 16 + half * 8);
    // B layout (32x16): lane half 0 -> K kk+[0..8)/kk+16+[0..8); half 1 -> K kk+8+[0..8)/kk+24+[0..8)
    b0v.h[0] = *(const v8h*)(B0 + kk + half * 8);
    b0v.h[1] = *(const v8h*)(B0 + kk + 16 + half * 8);
    b1v.h[0] = *(const v8h*)(B1 + kk + half * 8);
    b1v.h[1] = *(const v8h*)(B1 + kk + 16 + half * 8);
    c0 = __builtin_amdgcn_wmma_f32_16x16x32_f16(false, a.v, false, b0v.v,
                                                (short)0, c0, false, false);
    c1 = __builtin_amdgcn_wmma_f32_16x16x32_f16(false, a.v, false, b1v.v,
                                                (short)0, c1, false, false);
  }

#pragma unroll
  for (int r = 0; r < 8; ++r) {
    int row = M0 + r + 8 * half;
    X[(size_t)row * 256 + n0] = c0[r];
    X[(size_t)row * 256 + n1] = c1[r];
  }
}

// ---- edge phase 1: alpha logits + segment max (one wave per edge) ---------

__global__ __launch_bounds__(256)
void edge_alpha_max(const int* __restrict__ ei,
                    const float* __restrict__ XL, const float* __restrict__ XR,
                    const float* __restrict__ att,
                    float* __restrict__ alphaE, int* __restrict__ amaxI) {
  int wave = threadIdx.x >> 5, lane = threadIdx.x & 31;
  int e = blockIdx.x * 8 + wave;
  if (e >= ET_EDGES) return;
  int src = e < N_EDGES ? ei[e] : e - N_EDGES;
  int dst = e < N_EDGES ? ei[N_EDGES + e] : e - N_EDGES;

  int c0 = lane * 8;
  float l[8], r[8];
  *(float4*)&l[0] = *(const float4*)(XL + (size_t)src * 256 + c0);
  *(float4*)&l[4] = *(const float4*)(XL + (size_t)src * 256 + c0 + 4);
  *(float4*)&r[0] = *(const float4*)(XR + (size_t)dst * 256 + c0);
  *(float4*)&r[4] = *(const float4*)(XR + (size_t)dst * 256 + c0 + 4);

  float p = 0.f;
#pragma unroll
  for (int i = 0; i < 8; ++i) {
    float s = l[i] + r[i];
    s = s > 0.f ? s : 0.2f * s;        // leaky_relu(., 0.2)
    p += att[c0 + i] * s;              // att flat index == global channel
  }
  // reduce over the 8 lanes of each head (64 channels / 8 per lane)
  p += __shfl_xor(p, 1);
  p += __shfl_xor(p, 2);
  p += __shfl_xor(p, 4);
  if ((lane & 7) == 0) {
    int h = lane >> 3;
    alphaE[(size_t)e * 4 + h] = p;
    atomicMax(&amaxI[dst * 4 + h], fkey(p));
  }
}

// ---- edge phase 2: exp(alpha - max), segment sum ---------------------------

__global__ __launch_bounds__(256)
void edge_exp_den(const int* __restrict__ ei,
                  const float* __restrict__ alphaE, const int* __restrict__ amaxI,
                  float* __restrict__ exE, float* __restrict__ den) {
  int idx = blockIdx.x * blockDim.x + threadIdx.x;
  if (idx >= ET_EDGES * 4) return;
  int e = idx >> 2, h = idx & 3;
  int dst = e < N_EDGES ? ei[N_EDGES + e] : e - N_EDGES;
  float am = keyToFloat(amaxI[dst * 4 + h]);
  float ex = expf(alphaE[idx] - am);
  exE[idx] = ex;
  atomicAdd(&den[dst * 4 + h], ex);
}

// ---- edge phase 3: weighted aggregation (one wave per edge) ----------------

__global__ __launch_bounds__(256)
void edge_aggregate(const int* __restrict__ ei,
                    const float* __restrict__ XL,
                    const float* __restrict__ exE, const float* __restrict__ den,
                    float* __restrict__ agg) {
  int wave = threadIdx.x >> 5, lane = threadIdx.x & 31;
  int e = blockIdx.x * 8 + wave;
  if (e >= ET_EDGES) return;
  int src = e < N_EDGES ? ei[e] : e - N_EDGES;
  int dst = e < N_EDGES ? ei[N_EDGES + e] : e - N_EDGES;
  int h = lane >> 3;
  float a = exE[(size_t)e * 4 + h] / (den[dst * 4 + h] + 1e-16f);
  int c0 = lane * 8;
  const float* xl = XL + (size_t)src * 256 + c0;
  float* ag = agg + (size_t)dst * 256 + c0;
#pragma unroll
  for (int i = 0; i < 8; ++i) atomicAdd(&ag[i], xl[i] * a);
}

// ---- post: +bias, LN, ReLU (+ residual) for concat layers ------------------

__global__ __launch_bounds__(256)
void post_concat(const float* __restrict__ agg, const float* __restrict__ bias,
                 const float* __restrict__ g, const float* __restrict__ b,
                 const float* __restrict__ resid, float* __restrict__ out) {
  __shared__ float sbuf[256];
  int n = blockIdx.x, j = threadIdx.x;
  float v = agg[(size_t)n * 256 + j] + bias[j];
  float mean = block_reduce_sum(v, sbuf) * (1.f / 256);
  float d = v - mean;
  float var = block_reduce_sum(d * d, sbuf) * (1.f / 256);
  float y = d * rsqrtf(var + 1e-5f) * g[j] + b[j];
  y = y > 0.f ? y : 0.f;
  if (resid) y += resid[(size_t)n * 256 + j];
  out[(size_t)n * 256 + j] = y;
}

// ---- post: head mean, +bias, LN, ReLU for final layer ----------------------

__global__ __launch_bounds__(64)
void post_mean(const float* __restrict__ agg, const float* __restrict__ bias,
               const float* __restrict__ g, const float* __restrict__ b,
               float* __restrict__ out) {
  __shared__ float sbuf[64];
  int n = blockIdx.x, j = threadIdx.x;
  const float* a = agg + (size_t)n * 256;
  float v = 0.25f * (a[j] + a[64 + j] + a[128 + j] + a[192 + j]) + bias[j];
  float mean = block_reduce_sum(v, sbuf) * (1.f / 64);
  float d = v - mean;
  float var = block_reduce_sum(d * d, sbuf) * (1.f / 64);
  float y = d * rsqrtf(var + 1e-5f) * g[j] + b[j];
  out[(size_t)n * 64 + j] = y > 0.f ? y : 0.f;
}

// ---- predictor MLP + softplus ---------------------------------------------

__global__ __launch_bounds__(128)
void predictor_kernel(const int* __restrict__ node_idx, const int* __restrict__ eth,
                      const float* __restrict__ H3, const float* __restrict__ emb,
                      const float* __restrict__ W1, const float* __restrict__ b1,
                      const float* __restrict__ g1, const float* __restrict__ bb1,
                      const float* __restrict__ W2, const float* __restrict__ b2,
                      const float* __restrict__ g2, const float* __restrict__ bb2,
                      const float* __restrict__ W3, const float* __restrict__ b3,
                      float* __restrict__ out) {
  __shared__ float sz[96];
  __shared__ float s1[128];
  __shared__ float s2[64];
  __shared__ float sbuf[128];
  int bi = blockIdx.x, t = threadIdx.x;
  int ni = node_idx[bi], ee = eth[bi];
  if (t < 64) sz[t] = H3[(size_t)ni * 64 + t];
  else if (t < 96) sz[t] = emb[(size_t)ee * 32 + (t - 64)];
  __syncthreads();

  float a = b1[t];
  for (int k = 0; k < 96; ++k) a += sz[k] * W1[k * 128 + t];
  float mean = block_reduce_sum(a, sbuf) * (1.f / 128);
  float d = a - mean;
  float var = block_reduce_sum(d * d, sbuf) * (1.f / 128);
  float y = d * rsqrtf(var + 1e-5f) * g1[t] + bb1[t];
  s1[t] = y > 0.f ? y : 0.f;
  __syncthreads();

  float a2 = 0.f;
  if (t < 64) {
    a2 = b2[t];
    for (int k = 0; k < 128; ++k) a2 += s1[k] * W2[k * 64 + t];
  }
  float mean2 = block_reduce_sum(t < 64 ? a2 : 0.f, sbuf) * (1.f / 64);
  float d2 = t < 64 ? a2 - mean2 : 0.f;
  float var2 = block_reduce_sum(d2 * d2, sbuf) * (1.f / 64);
  if (t < 64) {
    float y2 = d2 * rsqrtf(var2 + 1e-5f) * g2[t] + bb2[t];
    s2[t] = y2 > 0.f ? y2 : 0.f;
  }
  __syncthreads();

  float p = block_reduce_sum(t < 64 ? s2[t] * W3[t] : 0.f, sbuf);
  if (t == 0) {
    p += b3[0];
    // stable softplus: max(x,0) + log1p(exp(-|x|))
    out[bi] = fmaxf(p, 0.f) + log1pf(expf(-fabsf(p)));
  }
}

// ---------------------------------------------------------------------------

extern "C" void kernel_launch(void* const* d_in, const int* in_sizes, int n_in,
                              void* d_out, int out_size, void* d_ws, size_t ws_size,
                              hipStream_t stream) {
  (void)in_sizes; (void)n_in; (void)out_size; (void)ws_size;

  const int*   eth    = (const int*)d_in[1];
  const int*   nidx   = (const int*)d_in[2];
  const float* census = (const float*)d_in[3];
  const float* pop    = (const float*)d_in[4];
  const int*   ei     = (const int*)d_in[5];

  const float* enc_W1 = (const float*)d_in[6];
  const float* enc_W2 = (const float*)d_in[7];
  const float* enc_b1 = (const float*)d_in[8];
  const float* enc_b2 = (const float*)d_in[9];
  const float* enc_ln1_b = (const float*)d_in[10];
  const float* enc_ln1_g = (const float*)d_in[11];
  const float* enc_ln2_b = (const float*)d_in[12];
  const float* enc_ln2_g = (const float*)d_in[13];
  const float* eth_emb = (const float*)d_in[14];
  const float* pred_W1 = (const float*)d_in[39];
  const float* pred_W2 = (const float*)d_in[40];
  const float* pred_W3 = (const float*)d_in[41];
  const float* pred_b1 = (const float*)d_in[42];
  const float* pred_b2 = (const float*)d_in[43];
  const float* pred_b3 = (const float*)d_in[44];
  const float* pred_ln1_b = (const float*)d_in[45];
  const float* pred_ln1_g = (const float*)d_in[46];
  const float* pred_ln2_b = (const float*)d_in[47];
  const float* pred_ln2_g = (const float*)d_in[48];

  float* ws = (float*)d_ws;
  // workspace layout (float offsets, all 64B aligned)
  float*     HA     = ws + 0;                    // 20000*256
  float*     HB     = ws + 5120000;              // 20000*256
  float*     XL     = ws + 10240000;             // 20000*256
  float*     XR     = ws + 15360000;             // 20000*256
  float*     agg    = ws + 20480000;             // 20000*256
  float*     alphaE = ws + 25600000;             // 340000*4
  float*     exE    = ws + 26960000;             // 340000*4
  int*       amaxI  = (int*)(ws + 28320000);     // 20000*4
  float*     den    = ws + 28400000;             // 20000*4
  _Float16*  Hh     = (_Float16*)(ws + 28480000);// 20000*256 halves
  _Float16*  WtL    = (_Float16*)(ws + 31040000);// 256*256 halves
  _Float16*  WtR    = (_Float16*)(ws + 31072768);// 256*256 halves
  // total: ~124.5 MB

  const int NEG_INF_KEY = (int)0x807FFFFF;       // fkey(-inf)

  // 1) encoder -> HA [N x 128]
  encoder_kernel<<<N_NODES, 128, 0, stream>>>(
      census, pop, enc_W1, enc_b1, enc_ln1_g, enc_ln1_b,
      enc_W2, enc_b2, enc_ln2_g, enc_ln2_b, HA);

  float* cur = HA;   // layer input
  float* nxt = HB;   // layer output
  const int Cins[3] = {128, 256, 256};

  for (int l = 0; l < 3; ++l) {
    const int Cin = Cins[l];
    const float* Wl   = (const float*)d_in[15 + 6 * l + 0];
    const float* Wr   = (const float*)d_in[15 + 6 * l + 1];
    const float* att  = (const float*)d_in[15 + 6 * l + 2];
    const float* bias = (const float*)d_in[15 + 6 * l + 3];
    const float* bl   = (const float*)d_in[15 + 6 * l + 4];
    const float* br   = (const float*)d_in[15 + 6 * l + 5];
    const float* nb   = (const float*)d_in[33 + 2 * l + 0];
    const float* ng   = (const float*)d_in[33 + 2 * l + 1];

    // f16 staging
    int nh = N_NODES * Cin;
    h_to_half<<<(nh + 255) / 256, 256, 0, stream>>>(cur, Hh, nh);
    transpose_w_f16<<<256, 256, 0, stream>>>(Wl, WtL, Cin);
    transpose_w_f16<<<256, 256, 0, stream>>>(Wr, WtR, Cin);

    // XL = H@Wl + bl ; XR = H@Wr + br  (WMMA)
    gat_gemm_f16<<<dim3(N_NODES / 16, 2), 256, 0, stream>>>(
        Hh, Cin, WtL, WtR, bl, br, XL, XR);

    // edge softmax + aggregation
    fill_i32<<<(N_NODES * 4 + 255) / 256, 256, 0, stream>>>(amaxI, NEG_INF_KEY, N_NODES * 4);
    fill_f32<<<(N_NODES * 4 + 255) / 256, 256, 0, stream>>>(den, 0.f, N_NODES * 4);
    fill_f32<<<(N_NODES * 256 + 255) / 256, 256, 0, stream>>>(agg, 0.f, N_NODES * 256);

    int eblocks = (ET_EDGES + 7) / 8;
    edge_alpha_max<<<eblocks, 256, 0, stream>>>(ei, XL, XR, att, alphaE, amaxI);
    edge_exp_den<<<(ET_EDGES * 4 + 255) / 256, 256, 0, stream>>>(ei, alphaE, amaxI, exE, den);
    edge_aggregate<<<eblocks, 256, 0, stream>>>(ei, XL, exE, den, agg);

    if (l < 2) {
      // residual only when shapes match and l>0  (layer 1)
      const float* resid = (l == 1) ? cur : nullptr;
      post_concat<<<N_NODES, 256, 0, stream>>>(agg, bias, ng, nb, resid, nxt);
    } else {
      post_mean<<<N_NODES, 64, 0, stream>>>(agg, bias, ng, nb, nxt);  // [N x 64]
    }
    float* t = cur; cur = nxt; nxt = t;
  }

  // after 3 swaps: cur holds final H3 [N x 64]
  predictor_kernel<<<BATCH, 128, 0, stream>>>(
      nidx, eth, cur, eth_emb,
      pred_W1, pred_b1, pred_ln1_g, pred_ln1_b,
      pred_W2, pred_b2, pred_ln2_g, pred_ln2_b,
      pred_W3, pred_b3, (float*)d_out);
}
